// ScannedRNN_57277683860104
// MI455X (gfx1250) — matrix operand order repaired and checked by reading, not compiled, not run on hardware
//
#include <hip/hip_runtime.h>
#include <hip/hip_bf16.h>
#include <cstdint>
#include <cstddef>

// Problem dimensions (from reference)
#define SEQ   512
#define BATCH 128
#define DIN   512
#define HDIM  512
#define G3    (3 * HDIM)   // 1536

typedef __attribute__((ext_vector_type(16))) __bf16 v16bf;
typedef __attribute__((ext_vector_type(8)))  __bf16 v8bf;
typedef __attribute__((ext_vector_type(8)))  float  v8f;

static __device__ __forceinline__ float sigmoidf_(float x) {
    return 1.0f / (1.0f + __expf(-x));
}

// ---------------------------------------------------------------------------
// Fragment loaders (CDNA5 WMMA 16x16x32 bf16 layouts, CDNA5 ISA §7.12.2)
//
// A (16x32, MxK): lane l -> row m = l&15, half h = l>>4.
//   elements 0..7  = A[m][k0 + h*8 .. +7]        (K 0..7 / 8..15)
//   elements 8..15 = A[m][k0 + 16 + h*8 .. +7]   (K 16..23 / 24..31)
//
// B (32x16, KxN): lane l -> col n = l&15, half h = l>>4.
//   element e = B[k0 + h*8 + e][n] (e<8), B[k0 + 16 + h*8 + e-8][n] (e>=8)
// B is pre-repacked so each lane's 16 elements are contiguous (two b128 loads).
//
// C/D (16x16 f32): lane l -> col n = l&15, group g = l>>4; vgpr v -> row g*8+v.
// ---------------------------------------------------------------------------

static __device__ __forceinline__ v16bf load_a_bf16(const __bf16* __restrict__ A,
                                                    int ld, int row0, int k0, int lane) {
    const int m = lane & 15, h = lane >> 4;
    const __bf16* p = A + (size_t)(row0 + m) * ld + k0 + h * 8;
    v8bf lo = *reinterpret_cast<const v8bf*>(p);
    v8bf hi = *reinterpret_cast<const v8bf*>(p + 16);
    v16bf a;
#pragma unroll
    for (int i = 0; i < 8; ++i) { a[i] = lo[i]; a[8 + i] = hi[i]; }
    return a;
}

// A fragment from f32 hidden state, applying per-row reset mask, cvt to bf16.
static __device__ __forceinline__ v16bf load_a_f32_masked(const float* __restrict__ A,
                                                          const uint8_t* __restrict__ resets,
                                                          int ld, int row0, int k0, int lane) {
    const int m = lane & 15, h = lane >> 4;
    const int row = row0 + m;
    const float keep = resets[row] ? 0.0f : 1.0f;
    const float* p = A + (size_t)row * ld + k0 + h * 8;
    v8f lo, hi;
#pragma unroll
    for (int i = 0; i < 8; ++i) { lo[i] = p[i]; hi[i] = p[16 + i]; }
    v16bf a;
#pragma unroll
    for (int i = 0; i < 8; ++i) {
        a[i]     = (__bf16)(lo[i] * keep);
        a[8 + i] = (__bf16)(hi[i] * keep);
    }
    return a;
}

// Repacked B fragment: tile (kt, nt); each lane owns 16 contiguous bf16.
static __device__ __forceinline__ v16bf load_b_packed(const __bf16* __restrict__ Bp,
                                                      int kt, int nt, int nTiles, int lane) {
    const __bf16* p = Bp + ((((size_t)kt * nTiles) + nt) * 32 + lane) * 16;
    v8bf lo = *reinterpret_cast<const v8bf*>(p);
    v8bf hi = *reinterpret_cast<const v8bf*>(p + 8);
    v16bf b;
#pragma unroll
    for (int i = 0; i < 8; ++i) { b[i] = lo[i]; b[8 + i] = hi[i]; }
    return b;
}

// ---------------------------------------------------------------------------
// Kernel 1: f32 -> bf16 elementwise convert (for `ins`)
// ---------------------------------------------------------------------------
__global__ void cvt_f32_bf16(const float* __restrict__ src, __bf16* __restrict__ dst, int n) {
    int i = blockIdx.x * blockDim.x + threadIdx.x;
    if (i < n) dst[i] = (__bf16)src[i];
}

// ---------------------------------------------------------------------------
// Kernel 2: convert + repack a [K, N] f32 weight into per-lane WMMA B-fragment
// order: dst[((kt*nTiles + nt)*32 + lane)*16 + e], lane = ((kk>>3)&1)*16 + n, e = (kk&7) + (kk>>4)*8.
// ---------------------------------------------------------------------------
__global__ void repack_w(const float* __restrict__ src, __bf16* __restrict__ dst,
                         int K, int N) {
    int idx = blockIdx.x * blockDim.x + threadIdx.x;
    if (idx >= K * N) return;
    int k = idx / N, c = idx % N;
    int kt = k >> 5, kk = k & 31;
    int nt = c >> 4, n = c & 15;
    int lane = (((kk >> 3) & 1) << 4) | n;
    int e = (kk & 7) + ((kk >> 4) << 3);
    int nTiles = N >> 4;
    size_t out = ((((size_t)kt * nTiles) + nt) * 32 + lane) * 16 + e;
    dst[out] = (__bf16)src[(size_t)k * N + c];
}

// ---------------------------------------------------------------------------
// Kernel 3: input_gates = ins @ W_in + b_in
//   A: [R=SEQ*BATCH, DIN] bf16 row-major; Bp: repacked [DIN, G3]; G: [R, G3] f32
//   Register-blocked: each wave computes a 32x64 macro-tile (2 M-tiles x 4
//   N-tiles = 8 accumulators); A fragments reused 4x, B fragments reused 2x.
//   grid = (G3/64, R/32), block = 32.
// ---------------------------------------------------------------------------
__global__ void __launch_bounds__(32)
gemm_input_gates(const __bf16* __restrict__ A, const __bf16* __restrict__ Bp,
                 const float* __restrict__ bias, float* __restrict__ G) {
    const int lane = threadIdx.x;
    const int nt0  = blockIdx.x * 4;     // first of 4 N tiles
    const int row0 = blockIdx.y * 32;    // first of 2 M tiles
    const int nTiles = G3 / 16;

    v8f acc[2][4];
#pragma unroll
    for (int i = 0; i < 2; ++i)
#pragma unroll
        for (int j = 0; j < 4; ++j) acc[i][j] = (v8f){};

#pragma unroll 4
    for (int kt = 0; kt < DIN / 32; ++kt) {
        v16bf a0 = load_a_bf16(A, DIN, row0,      kt * 32, lane);
        v16bf a1 = load_a_bf16(A, DIN, row0 + 16, kt * 32, lane);
        v16bf b[4];
#pragma unroll
        for (int j = 0; j < 4; ++j) b[j] = load_b_packed(Bp, kt, nt0 + j, nTiles, lane);
#pragma unroll
        for (int j = 0; j < 4; ++j) {
            acc[0][j] = __builtin_amdgcn_wmma_f32_16x16x32_bf16(false, a0, false, b[j],
                                                                (short)0, acc[0][j], false, false);
            acc[1][j] = __builtin_amdgcn_wmma_f32_16x16x32_bf16(false, a1, false, b[j],
                                                                (short)0, acc[1][j], false, false);
        }
    }

    const int n = lane & 15, group = lane >> 4;
#pragma unroll
    for (int j = 0; j < 4; ++j) {
        const int col = (nt0 + j) * 16 + n;
        const float bv = bias[col];
#pragma unroll
        for (int i = 0; i < 2; ++i) {
#pragma unroll
            for (int v = 0; v < 8; ++v) {
                int row = row0 + i * 16 + group * 8 + v;
                G[(size_t)row * G3 + col] = acc[i][j][v] + bv;
            }
        }
    }
}

// ---------------------------------------------------------------------------
// Kernel 4: one GRU scan step.
//   hidden_gates = mask(h_prev) @ W_h  (three 16x16 gate tiles per wave, shared A)
//   then z/r/n gating; writes h_new. grid = (HDIM/16, BATCH/16), block = 32.
// ---------------------------------------------------------------------------
__global__ void __launch_bounds__(32)
gru_step(const float* __restrict__ h_prev,      // [BATCH, HDIM]
         const float* __restrict__ Gs,          // [BATCH, G3] input gates for step s
         const uint8_t* __restrict__ resets_s,  // [BATCH]
         const __bf16* __restrict__ Whp,        // repacked [HDIM, G3]
         float* __restrict__ h_out) {           // [BATCH, HDIM]
    const int lane = threadIdx.x;
    const int jT   = blockIdx.x;          // N tile within one gate (0..31)
    const int col0 = jT * 16;
    const int row0 = blockIdx.y * 16;
    const int nTiles = G3 / 16;           // 96: z tiles 0..31, r 32..63, n 64..95

    v8f accz = {}, accr = {}, accn = {};
#pragma unroll 4
    for (int kt = 0; kt < HDIM / 32; ++kt) {
        v16bf a  = load_a_f32_masked(h_prev, resets_s, HDIM, row0, kt * 32, lane);
        v16bf bz = load_b_packed(Whp, kt, jT,      nTiles, lane);
        v16bf br = load_b_packed(Whp, kt, jT + 32, nTiles, lane);
        v16bf bn = load_b_packed(Whp, kt, jT + 64, nTiles, lane);
        accz = __builtin_amdgcn_wmma_f32_16x16x32_bf16(false, a, false, bz,
                                                       (short)0, accz, false, false);
        accr = __builtin_amdgcn_wmma_f32_16x16x32_bf16(false, a, false, br,
                                                       (short)0, accr, false, false);
        accn = __builtin_amdgcn_wmma_f32_16x16x32_bf16(false, a, false, bn,
                                                       (short)0, accn, false, false);
    }

    const int n = lane & 15, group = lane >> 4;
#pragma unroll
    for (int v = 0; v < 8; ++v) {
        const int row = row0 + group * 8 + v;
        const int col = col0 + n;
        const size_t gbase = (size_t)row * G3 + col;
        float gz = Gs[gbase];
        float gr = Gs[gbase + HDIM];
        float gn = Gs[gbase + 2 * HDIM];
        float z  = sigmoidf_(gz + accz[v]);
        float r  = sigmoidf_(gr + accr[v]);
        float nn = tanhf(gn + r * accn[v]);
        float hp = resets_s[row] ? 0.0f : h_prev[(size_t)row * HDIM + col];
        h_out[(size_t)row * HDIM + col] = (1.0f - z) * nn + z * hp;
    }
}

// ---------------------------------------------------------------------------
// Host launcher
// Inputs (setup_inputs order): carry f32[128,512], ins f32[512,128,512],
// resets bool[512,128], W_in f32[512,1536], b_in f32[1536], W_h f32[512,1536]
// Output: final_carry f32[128,512] ++ outputs f32[512,128,512]
// ---------------------------------------------------------------------------
extern "C" void kernel_launch(void* const* d_in, const int* in_sizes, int n_in,
                              void* d_out, int out_size, void* d_ws, size_t ws_size,
                              hipStream_t stream) {
    const float*   carry  = (const float*)d_in[0];
    const float*   ins    = (const float*)d_in[1];
    const uint8_t* resets = (const uint8_t*)d_in[2];   // jax bool = 1 byte
    const float*   W_in   = (const float*)d_in[3];
    const float*   b_in   = (const float*)d_in[4];
    const float*   W_h    = (const float*)d_in[5];

    float* out_carry = (float*)d_out;                       // [BATCH*HDIM]
    float* outputs   = (float*)d_out + BATCH * HDIM;        // [SEQ][BATCH*HDIM]

    // Workspace carving (256B aligned)
    char* ws = (char*)d_ws;
    auto align256 = [](size_t x) { return (x + 255) & ~(size_t)255; };
    const size_t R = (size_t)SEQ * BATCH;

    __bf16* insb = (__bf16*)ws;                  ws += align256(R * DIN * sizeof(__bf16));
    __bf16* Winp = (__bf16*)ws;                  ws += align256((size_t)DIN * G3 * sizeof(__bf16));
    __bf16* Whp  = (__bf16*)ws;                  ws += align256((size_t)HDIM * G3 * sizeof(__bf16));
    float*  Gw   = (float*)ws;                   /* R * G3 f32 */

    // 1) convert ins to bf16
    {
        int n = (int)(R * DIN);
        cvt_f32_bf16<<<(n + 255) / 256, 256, 0, stream>>>(ins, insb, n);
    }
    // 2) convert + repack weights into WMMA B-fragment order
    {
        int n = DIN * G3;
        repack_w<<<(n + 255) / 256, 256, 0, stream>>>(W_in, Winp, DIN, G3);
        repack_w<<<(n + 255) / 256, 256, 0, stream>>>(W_h,  Whp,  HDIM, G3);
    }
    // 3) big GEMM: all input-side gates (32x64 macro-tile per wave)
    {
        dim3 grid(G3 / 64, (unsigned)(R / 32));   // 24 x 2048 macro-tiles
        gemm_input_gates<<<grid, 32, 0, stream>>>(insb, Winp, b_in, Gw);
    }
    // 4) sequential scan: 512 step kernels (graph-captured, replayed as a chain)
    {
        dim3 grid(HDIM / 16, BATCH / 16);         // 32 x 8 tiles, 1 wave each
        for (int s = 0; s < SEQ; ++s) {
            const float* hp = (s == 0) ? carry : (outputs + (size_t)(s - 1) * BATCH * HDIM);
            gru_step<<<grid, 32, 0, stream>>>(
                hp,
                Gw + (size_t)s * BATCH * G3,
                resets + (size_t)s * BATCH,
                Whp,
                outputs + (size_t)s * BATCH * HDIM);
        }
    }
    // 5) final carry = outputs[SEQ-1]
    hipMemcpyAsync(out_carry, outputs + (size_t)(SEQ - 1) * BATCH * HDIM,
                   (size_t)BATCH * HDIM * sizeof(float),
                   hipMemcpyDeviceToDevice, stream);
}
